// Quantizer_84138409328640
// MI455X (gfx1250) — matrix operand order repaired and checked by reading, not compiled, not run on hardware
//
#include <hip/hip_runtime.h>

// ---------------------------------------------------------------------------
// VQ-VAE quantizer for gfx1250 (MI455X).
//   score(n,k) = ||e_k||^2 - 2 x_n.e_k   (row constant dropped for argmin)
// GEMM: v_wmma_f32_16x16x32_bf16, 3-term bf16 error compensation
//   s = xh.eh + xh.el + xl.eh   (~fp32 accuracy at bf16 speed).
// A (x rows) persistent in VGPR fragments; B (codes) streamed to LDS with
// global_load_async_to_lds_b128 (ASYNCcnt) in a double buffer.
// ---------------------------------------------------------------------------

typedef __attribute__((ext_vector_type(16))) __bf16 v16bf;
typedef __attribute__((ext_vector_type(8)))  float  v8f;

union Pack8 { __bf16 b[8]; uint4 u; };

static constexpr int DD   = 256;    // feature dim
static constexpr int KK   = 8192;   // number of codes
static constexpr int NN   = 16384;  // rows
static constexpr int MT   = 128;    // rows per workgroup (8 waves x 16 rows)
static constexpr int KC   = 32;     // codes per chunk
static constexpr int SA   = 264;    // padded LDS stride (elems) = 528B = 33*16B
static constexpr int NCHUNK = KK / KC;  // 256

static constexpr unsigned BBYTES  = KC * SA * 2;   // 16896 bytes per matrix buf
static constexpr unsigned OFF_B0H = 0;
static constexpr unsigned OFF_B0L = BBYTES;
static constexpr unsigned OFF_B1H = 2 * BBYTES;
static constexpr unsigned OFF_B1L = 3 * BBYTES;
static constexpr unsigned LDS_BYTES = 4 * BBYTES;  // ~66 KB

__device__ __forceinline__ v16bf ld_frag(const __bf16* p0, const __bf16* p1) {
  union { uint4 u[2]; v16bf v; } t;
  t.u[0] = *(const uint4*)p0;
  t.u[1] = *(const uint4*)p1;
  return t.v;
}

__device__ __forceinline__ v8f wmma_bf16(v16bf a, v16bf b, v8f c) {
  return __builtin_amdgcn_wmma_f32_16x16x32_bf16(
      false, a, false, b, (short)0, c, false, false);
}

// Async DMA: 16B from global (saddr + voffset) straight into LDS byte offset.
__device__ __forceinline__ void async_b128(unsigned lds_off, unsigned voff,
                                           const void* base) {
  asm volatile("global_load_async_to_lds_b128 %0, %1, %2"
               :: "v"(lds_off), "v"(voff), "s"(base)
               : "memory");
}

__device__ __forceinline__ void wait_async0() {
  asm volatile("s_wait_asynccnt 0x0" ::: "memory");
}

// --- prep: transpose codebook to [k][d], split bf16 hi/lo, compute ||e_k||^2
__global__ __launch_bounds__(256) void prep_embed(
    const float* __restrict__ emb, __bf16* __restrict__ ehi,
    __bf16* __restrict__ elo, float* __restrict__ eT, float* __restrict__ ssq) {
  const int k = blockIdx.x * 256 + threadIdx.x;  // coalesced over k for each d
  float acc = 0.f;
  for (int d0 = 0; d0 < DD; d0 += 8) {
    float v[8];
#pragma unroll
    for (int j = 0; j < 8; ++j) {
      float x = emb[(size_t)(d0 + j) * KK + k];
      v[j] = x;
      acc += x * x;
    }
    Pack8 h, l;
#pragma unroll
    for (int j = 0; j < 8; ++j) {
      __bf16 hb = (__bf16)v[j];
      h.b[j] = hb;
      l.b[j] = (__bf16)(v[j] - (float)hb);
    }
    *(uint4*)(ehi + (size_t)k * DD + d0) = h.u;
    *(uint4*)(elo + (size_t)k * DD + d0) = l.u;
    *(float4*)(eT + (size_t)k * DD + d0)     = make_float4(v[0], v[1], v[2], v[3]);
    *(float4*)(eT + (size_t)k * DD + d0 + 4) = make_float4(v[4], v[5], v[6], v[7]);
  }
  ssq[k] = acc;
}

// --- prep: split x into bf16 hi/lo (fully coalesced)
__global__ __launch_bounds__(256) void prep_x(
    const float* __restrict__ x, __bf16* __restrict__ xhi, __bf16* __restrict__ xlo) {
  const size_t i = ((size_t)blockIdx.x * 256 + threadIdx.x) * 8;
  float4 a = *(const float4*)(x + i);
  float4 b = *(const float4*)(x + i + 4);
  float v[8] = {a.x, a.y, a.z, a.w, b.x, b.y, b.z, b.w};
  Pack8 h, l;
#pragma unroll
  for (int j = 0; j < 8; ++j) {
    __bf16 hb = (__bf16)v[j];
    h.b[j] = hb;
    l.b[j] = (__bf16)(v[j] - (float)hb);
  }
  *(uint4*)(xhi + i) = h.u;
  *(uint4*)(xlo + i) = l.u;
}

// --- main: fused bf16x3 WMMA GEMM + running argmin. Never materializes dist.
__global__ __launch_bounds__(256) void vq_argmin(
    const __bf16* __restrict__ xhi, const __bf16* __restrict__ xlo,
    const __bf16* __restrict__ ehi, const __bf16* __restrict__ elo,
    const float* __restrict__ ssq, int* __restrict__ idx_out) {
  extern __shared__ char smem[];

  const int tid  = threadIdx.x;
  const int wave = tid >> 5;
  const int lane = tid & 31;
  const int half = lane >> 4;   // 0: lanes 0-15, 1: lanes 16-31
  const int nl   = lane & 15;
  const int row0 = blockIdx.x * MT;

  // LDS raw base offset of the dynamic allocation (for async DMA dest).
  const unsigned ldsbase = __builtin_amdgcn_groupstaticsize();

  // Per-thread async segment map: 1024 16B segs per matrix -> 4 per thread.
  unsigned gsegv[4];  // global voffset (bytes) within a code-chunk
  unsigned lsegv[4];  // LDS offset (bytes) within a buffer
  unsigned segk[4];   // code index within chunk
#pragma unroll
  for (int u = 0; u < 4; ++u) {
    const int i = tid + u * 256;
    const int j = i >> 5;            // code within chunk (0..31)
    const int db = (i & 31) * 16;    // byte offset within the 512B code row
    segk[u]  = (unsigned)j;
    gsegv[u] = (unsigned)db;         // + (kbase + j)*512 at issue time
    lsegv[u] = (unsigned)(j * (SA * 2) + db);
  }

  // Persistent A fragments: wave's 16 rows x 256 dims, hi+lo (128 VGPRs).
  v16bf a_h[DD / 32], a_l[DD / 32];
  {
    const __bf16* xr  = xhi + (size_t)(row0 + wave * 16 + nl) * DD + half * 16;
    const __bf16* xrl = xlo + (size_t)(row0 + wave * 16 + nl) * DD + half * 16;
#pragma unroll
    for (int s = 0; s < DD / 32; ++s) {
      a_h[s] = ld_frag(xr  + s * 32, xr  + s * 32 + 8);
      a_l[s] = ld_frag(xrl + s * 32, xrl + s * 32 + 8);
    }
  }

  // Prologue: async-load chunk 0 into buffer 0.
#pragma unroll
  for (int u = 0; u < 4; ++u) {
    const unsigned gv = segk[u] * 512u + gsegv[u];
    async_b128(ldsbase + OFF_B0H + lsegv[u], gv, ehi);
    async_b128(ldsbase + OFF_B0L + lsegv[u], gv, elo);
  }

  float minv[8];
  int   mini[8];
#pragma unroll
  for (int r = 0; r < 8; ++r) { minv[r] = 3.4e38f; mini[r] = 0; }

  const unsigned fragoff = (unsigned)((nl * SA + half * 16) * 2);  // bytes

  for (int c = 0; c < NCHUNK; ++c) {
    const int kbase = c * KC;

    // Own async loads for chunk c landed; barrier makes the whole buffer
    // valid for every wave AND proves nobody still reads buffer c^1.
    wait_async0();
    __syncthreads();

    // Kick chunk c+1 into the other buffer; DMA overlaps the WMMA block.
    if (c + 1 < NCHUNK) {
      const unsigned kb = (unsigned)(kbase + KC);
      const unsigned oH = ldsbase + ((c & 1) ? OFF_B0H : OFF_B1H);
      const unsigned oL = oH + BBYTES;
#pragma unroll
      for (int u = 0; u < 4; ++u) {
        const unsigned gv = (kb + segk[u]) * 512u + gsegv[u];
        async_b128(oH + lsegv[u], gv, ehi);
        async_b128(oL + lsegv[u], gv, elo);
      }
    }

    const __bf16* Bh = (const __bf16*)(smem + ((c & 1) ? OFF_B1H : OFF_B0H));
    const __bf16* Bl = Bh + KC * SA;  // lo buffer is contiguous after hi
    const __bf16* b_r0h = (const __bf16*)((const char*)Bh + fragoff);
    const __bf16* b_r0l = (const __bf16*)((const char*)Bl + fragoff);
    const __bf16* b_r1h = b_r0h + 16 * SA;  // code tile 1 (codes 16..31)
    const __bf16* b_r1l = b_r0l + 16 * SA;

    v8f acc0 = {}, acc1 = {};
#pragma unroll
    for (int s = 0; s < DD / 32; ++s) {
      const int d0 = s * 32;
      v16bf b0h = ld_frag(b_r0h + d0, b_r0h + d0 + 8);
      v16bf b0l = ld_frag(b_r0l + d0, b_r0l + d0 + 8);
      acc0 = wmma_bf16(a_h[s], b0h, acc0);
      acc0 = wmma_bf16(a_h[s], b0l, acc0);
      acc0 = wmma_bf16(a_l[s], b0h, acc0);
      v16bf b1h = ld_frag(b_r1h + d0, b_r1h + d0 + 8);
      v16bf b1l = ld_frag(b_r1l + d0, b_r1l + d0 + 8);
      acc1 = wmma_bf16(a_h[s], b1h, acc1);
      acc1 = wmma_bf16(a_h[s], b1l, acc1);
      acc1 = wmma_bf16(a_l[s], b1h, acc1);
    }

    // score = ||e||^2 - 2*dot ; strict '<' keeps lowest index (k ascending).
    {
      const int k0 = kbase + nl;
      const float s0 = ssq[k0];
      const float s1 = ssq[k0 + 16];
#pragma unroll
      for (int r = 0; r < 8; ++r) {
        float d0v = s0 - 2.f * acc0[r];
        if (d0v < minv[r]) { minv[r] = d0v; mini[r] = k0; }
        float d1v = s1 - 2.f * acc1[r];
        if (d1v < minv[r]) { minv[r] = d1v; mini[r] = k0 + 16; }
      }
    }
  }

  // Cross-lane argmin over the 16 lanes holding each row (C layout:
  // lane = column, VGPR r = row 8*half + r). Lexicographic tie-break.
#pragma unroll
  for (int r = 0; r < 8; ++r) {
    float v = minv[r];
    int   i = mini[r];
#pragma unroll
    for (int m = 8; m >= 1; m >>= 1) {
      float ov = __shfl_xor(v, m, 32);
      int   oi = __shfl_xor(i, m, 32);
      if (ov < v || (ov == v && oi < i)) { v = ov; i = oi; }
    }
    if (nl == 0) idx_out[row0 + wave * 16 + half * 8 + r] = i;
  }
}

// --- gather winning code (coalesced via eT) + per-row squared error
__global__ __launch_bounds__(256) void gather_loss(
    const float* __restrict__ x, const float* __restrict__ eT,
    const int* __restrict__ idx, float* __restrict__ out,
    float* __restrict__ partial) {
  __shared__ float red[256];
  const int row = blockIdx.x;
  const int d = threadIdx.x;
  const int k = idx[row];
  const float q  = eT[(size_t)k * DD + d];
  const float xv = x[(size_t)row * DD + d];
  out[(size_t)row * DD + d] = q;  // q_st == q numerically
  const float diff = q - xv;
  red[d] = diff * diff;
  __syncthreads();
  for (int s = 128; s > 0; s >>= 1) {
    if (d < s) red[d] += red[d + s];
    __syncthreads();
  }
  if (d == 0) partial[row] = red[0];
}

// --- deterministic final reduction: loss = 2 * sum / (N*D)
__global__ __launch_bounds__(256) void final_loss(
    const float* __restrict__ partial, float* __restrict__ out_loss) {
  __shared__ float red[256];
  const int t = threadIdx.x;
  float a = 0.f;
  for (int i = t; i < NN; i += 256) a += partial[i];
  red[t] = a;
  __syncthreads();
  for (int s = 128; s > 0; s >>= 1) {
    if (t < s) red[t] += red[t + s];
    __syncthreads();
  }
  if (t == 0) out_loss[0] = 2.0f * red[0] / (float)((size_t)NN * DD);
}

extern "C" void kernel_launch(void* const* d_in, const int* in_sizes, int n_in,
                              void* d_out, int out_size, void* d_ws, size_t ws_size,
                              hipStream_t stream) {
  const float* inpt = (const float*)d_in[0];  // [16,1024,256] fp32
  const float* emb  = (const float*)d_in[1];  // [256,8192]    fp32
  float* out = (float*)d_out;                 // 16*1024*256 q_st + 1 loss

  char* ws = (char*)d_ws;
  __bf16* xhi = (__bf16*)(ws);                                  // 8 MB
  __bf16* xlo = (__bf16*)(ws + (8ull  << 20));                  // 8 MB
  __bf16* ehi = (__bf16*)(ws + (16ull << 20));                  // 4 MB
  __bf16* elo = (__bf16*)(ws + (20ull << 20));                  // 4 MB
  float*  eT  = (float*) (ws + (24ull << 20));                  // 8 MB
  float*  ssq = (float*) (ws + (32ull << 20));                  // 32 KB
  int*    idx = (int*)   (ws + (32ull << 20) + (64ull << 10));  // 64 KB
  float*  prt = (float*) (ws + (32ull << 20) + (192ull << 10)); // 64 KB

  prep_embed<<<KK / 256, 256, 0, stream>>>(emb, ehi, elo, eT, ssq);
  prep_x<<<(NN * DD) / (256 * 8), 256, 0, stream>>>(inpt, xhi, xlo);

  vq_argmin<<<NN / MT, 256, LDS_BYTES, stream>>>(xhi, xlo, ehi, elo, ssq, idx);

  gather_loss<<<NN, 256, 0, stream>>>(inpt, eT, idx, out, prt);
  final_loss<<<1, 256, 0, stream>>>(prt, out + (size_t)NN * DD);
}
